// HGAN_72765335929015
// MI455X (gfx1250) — compile-verified, matrix operand order checked by reading, not run on hardware
//
#include <hip/hip_runtime.h>
#include <hip/hip_bf16.h>
#include <math.h>

// Problem constants (reference: B,M,O,D,H,DK,HID,L = 8,64,512,256,8,32,512,3)
#define BB   8
#define MMm  64
#define OOo  512
#define DDd  256
#define HHh  8
#define DKk  32
#define RARC (BB*MMm*OOo)   // 262144 rows of h_arc
#define ATT_SCALE 0.0625f   // 1/sqrt(D) = 1/16

typedef __attribute__((ext_vector_type(16))) _Float16 v16h;
typedef __attribute__((ext_vector_type(8)))  _Float16 v8h;
typedef __attribute__((ext_vector_type(8)))  float    v8f;
typedef __attribute__((ext_vector_type(4)))  float    v4f;

#define LDST 48  // LDS row stride in halfs (96B: 16B-aligned rows, staggers banks)

// GEMM config bits (compile-time)
#define CFG_CS   1   // fold per-column rmsnorm scale into A
#define CFG_OUTH 2   // fp16 output
#define CFG_GELU 4   // exact gelu epilogue
#define CFG_R1   8   // + residual 1
#define CFG_R2   16  // + residual 2
#define CFG_BIAS 32  // + bias

// Load one 16x32 fp16 fragment from an LDS tile (row-major, stride LDST halfs)
// per CDNA5 ISA 7.12.2 "16-bit A-Matrix 16x32":
//  lane l (r = l&15, hi = l>>4) holds row r, K = {hi*8..+7} U {16+hi*8..+7}.
// B fragments use the same pattern with the tile stored column-major (Wt[n][k]).
__device__ __forceinline__ v16h frag_ld(const _Float16* base, int lane) {
  const int r = lane & 15, hi = lane >> 4;
  const _Float16* p = base + r * LDST + hi * 8;
  v8h lo = *(const v8h*)(p);        // ds_load_b128
  v8h hh = *(const v8h*)(p + 16);   // ds_load_b128
  v16h o;
#pragma unroll
  for (int i = 0; i < 8; ++i) { o[i] = lo[i]; o[8 + i] = hh[i]; }
  return o;
}

// Weight prep: W (fp32, K x N row-major, optional batch) -> Wt (fp16, N x K).
// Small matrices, done once per GEMM; lets the GEMM stage B with pure b128 traffic.
__global__ void hgan_prep_w(const float* __restrict__ W, _Float16* __restrict__ Wt,
                            int K, int N, long sW) {
  const long kn = (long)K * N;
  long i = (long)blockIdx.x * blockDim.x + threadIdx.x;
  if (i >= kn) return;
  const long z = blockIdx.y;
  const int n = (int)(i / K), k = (int)(i % K);
  Wt[z * kn + i] = (_Float16)W[z * sW + (long)k * N + n];
}

// Fused GEMM: out = act( (rmsnorm-folded A) @ W + bias ) [+res1][+res2]
//  A fp32 RxK row-major; per-row invnorm + per-col scale folded in (fused rmsnorm).
//  Wt fp16 NxK (pre-transposed). 64x64 tile, 128 thr = 4 waves, 2x2 wmma_f32_16x16x32_f16
//  per wave. Double-buffered LDS: fetch tile k+1 into regs while WMMAs consume tile k.
template <int CFG>
__global__ __launch_bounds__(128) void hgan_wmma_gemm(
    const float* __restrict__ A, const float* __restrict__ invn,
    const float* __restrict__ cs, const _Float16* __restrict__ Wt,
    const float* __restrict__ bias, float* __restrict__ outf,
    _Float16* __restrict__ outh, const float* __restrict__ res1,
    const float* __restrict__ res2, int K, int N,
    long sA, long sW, long sOut)
{
  constexpr bool CS   = CFG & CFG_CS;
  constexpr bool OUTH = CFG & CFG_OUTH;
  constexpr bool GELU = CFG & CFG_GELU;
  constexpr bool R1   = CFG & CFG_R1;
  constexpr bool R2   = CFG & CFG_R2;
  constexpr bool BIAS = CFG & CFG_BIAS;

  __shared__ __align__(16) _Float16 Al[2][64 * LDST];
  __shared__ __align__(16) _Float16 Wl[2][64 * LDST];  // column-major: Wl[n][k]
  const int tid = threadIdx.x, lane = tid & 31, wv = tid >> 5;
  const int wr = (wv & 1) * 32, wc = (wv >> 1) * 32;
  const int row0 = blockIdx.y * 64, n0 = blockIdx.x * 64;
  const long zA = (long)blockIdx.z * sA;
  const long zW = (long)blockIdx.z * sW;
  const long zO = (long)blockIdx.z * sOut;
  v8f acc00 = {}, acc01 = {}, acc10 = {}, acc11 = {};

  // Staging split: each thread owns 16 contiguous elements of the 64x32 tile.
  const int ar = tid >> 1, akb = (tid & 1) << 4;   // A: row ar, k sub-offset akb
  const float ainv = invn ? invn[row0 + ar] : 1.0f;
  const float* abase = A + zA + (long)(row0 + ar) * K + akb;
  const _Float16* wbase = Wt + zW + (long)(n0 + ar) * K + akb;  // same split for Wt

  // Fetch one 64x32 A-tile slice + 64x32 Wt-tile slice into registers (all b128).
  auto fetch = [&](int k0, v8h& a0, v8h& a1, v8h& w0, v8h& w1) {
    v4f t[4];
    const v4f* a4 = (const v4f*)(abase + k0);
#pragma unroll
    for (int i = 0; i < 4; ++i) t[i] = a4[i];
    if (CS) {
      const v4f* c4 = (const v4f*)(cs + k0 + akb);
#pragma unroll
      for (int i = 0; i < 4; ++i) t[i] *= c4[i];
    }
#pragma unroll
    for (int i = 0; i < 4; ++i) t[i] *= ainv;
#pragma unroll
    for (int i = 0; i < 8; ++i) {
      a0[i] = (_Float16)t[i >> 2][i & 3];
      a1[i] = (_Float16)t[2 + (i >> 2)][i & 3];
    }
    const v8h* w8 = (const v8h*)(wbase + k0);
    w0 = w8[0]; w1 = w8[1];
  };
  auto stage = [&](int p, v8h a0, v8h a1, v8h w0, v8h w1) {
    *(v8h*)&Al[p][ar * LDST + akb]     = a0;
    *(v8h*)&Al[p][ar * LDST + akb + 8] = a1;
    *(v8h*)&Wl[p][ar * LDST + akb]     = w0;
    *(v8h*)&Wl[p][ar * LDST + akb + 8] = w1;
  };

  {
    v8h a0, a1, w0, w1;
    fetch(0, a0, a1, w0, w1);
    stage(0, a0, a1, w0, w1);
  }
  __syncthreads();
  int p = 0;
  for (int k0 = 0; k0 < K; k0 += 32) {
    v8h a0, a1, w0, w1;
    const bool more = (k0 + 32) < K;
    if (more) fetch(k0 + 32, a0, a1, w0, w1);   // overlap with WMMA below

    v16h fa0 = frag_ld(&Al[p][(wr)      * LDST], lane);
    v16h fa1 = frag_ld(&Al[p][(wr + 16) * LDST], lane);
    v16h fb0 = frag_ld(&Wl[p][(wc)      * LDST], lane);
    v16h fb1 = frag_ld(&Wl[p][(wc + 16) * LDST], lane);
    acc00 = __builtin_amdgcn_wmma_f32_16x16x32_f16(false, fa0, false, fb0, (short)0, acc00, false, false);
    acc01 = __builtin_amdgcn_wmma_f32_16x16x32_f16(false, fa0, false, fb1, (short)0, acc01, false, false);
    acc10 = __builtin_amdgcn_wmma_f32_16x16x32_f16(false, fa1, false, fb0, (short)0, acc10, false, false);
    acc11 = __builtin_amdgcn_wmma_f32_16x16x32_f16(false, fa1, false, fb1, (short)0, acc11, false, false);

    if (more) stage(p ^ 1, a0, a1, w0, w1);
    __syncthreads();
    p ^= 1;
  }

  // Epilogue. C/D layout: lane l -> n = l&15; VGPR j -> m = j + 8*(l>>4).
  const int nl = lane & 15, hi = lane >> 4;
  float bva[2] = {0.f, 0.f};
  if (BIAS) {
    bva[0] = bias[n0 + wc + nl];
    bva[1] = bias[n0 + wc + 16 + nl];
  }
#pragma unroll
  for (int rt = 0; rt < 2; ++rt) {
#pragma unroll
    for (int ct = 0; ct < 2; ++ct) {
      v8f accv = (rt == 0) ? (ct == 0 ? acc00 : acc01) : (ct == 0 ? acc10 : acc11);
      const long obase = zO + (long)(row0 + wr + rt * 16 + hi * 8) * N
                       + (n0 + wc + ct * 16 + nl);
      float r1v[8], r2v[8];
      if (R1) {
#pragma unroll
        for (int j = 0; j < 8; ++j) r1v[j] = res1[obase + (long)j * N];
      }
      if (R2) {
#pragma unroll
        for (int j = 0; j < 8; ++j) r2v[j] = res2[obase + (long)j * N];
      }
#pragma unroll
      for (int j = 0; j < 8; ++j) {
        float cv = accv[j];
        if (BIAS) cv += bva[ct];
        if (GELU) cv = 0.5f * cv * (1.0f + erff(cv * 0.70710678118f)); // exact gelu
        if (R1) cv += r1v[j];
        if (R2) cv += r2v[j];
        if (OUTH) outh[obase + (long)j * N] = (_Float16)cv;
        else      outf[obase + (long)j * N] = cv;
      }
    }
  }
}

// Per-row 1/sqrt(mean(x^2)+eps); one wave32 per row, 8 contiguous floats per lane (b128s).
__global__ __launch_bounds__(256) void hgan_rowinv(
    const float* __restrict__ x, float* __restrict__ inv, int R, int K)
{
  const int row = blockIdx.x * 8 + (threadIdx.x >> 5);
  const int lane = threadIdx.x & 31;
  if (row >= R) return;
  const float* p = x + (long)row * K;
  float s = 0.f;
  for (int base = 0; base < K; base += 256) {
    const v4f* p4 = (const v4f*)(p + base + lane * 8);
    v4f u0 = p4[0], u1 = p4[1];
#pragma unroll
    for (int i = 0; i < 4; ++i) s += u0[i] * u0[i] + u1[i] * u1[i];
  }
#pragma unroll
  for (int off = 16; off > 0; off >>= 1) s += __shfl_xor(s, off, 32);
  if (lane == 0) inv[row] = rsqrtf(s / (float)K + 1e-8f);
}

__global__ void hgan_i2f(const int* __restrict__ a, float* __restrict__ o, int n) {
  int i = blockIdx.x * blockDim.x + threadIdx.x;
  if (i < n) o[i] = (float)a[i];
}

__device__ __forceinline__ float dot32_hq(const _Float16* kp, const float* qs) {
  const v8h* k8 = (const v8h*)kp;
  float l = 0.f;
#pragma unroll
  for (int i = 0; i < 4; ++i) {
    v8h hv = k8[i];
#pragma unroll
    for (int j = 0; j < 8; ++j) l += (float)hv[j] * qs[i * 8 + j];
  }
  return l;
}

// Machine-layer attention: per (b,m): logits over o, max(o), mask, softmax(o), sum_o p*v.
__global__ __launch_bounds__(256) void hgan_attn_machine(
    const float* __restrict__ q, const _Float16* __restrict__ kh,
    const _Float16* __restrict__ vh, const int* __restrict__ msk,
    float* __restrict__ res)
{
  __shared__ float qs[32];
  __shared__ float p[OOo];
  __shared__ float red[256];
  __shared__ float part[8][32];
  const int bm = blockIdx.x;                 // b*M + m
  const int tid = threadIdx.x;
  const long rowb = (long)bm * OOo;          // (b*M+m)*O
  const int o0 = tid, o1 = tid + 256;
  for (int h = 0; h < HHh; ++h) {
    if (tid < 32) qs[tid] = q[(long)bm * DDd + h * DKk + tid];
    __syncthreads();
    float l0 = dot32_hq(kh + (rowb + o0) * DDd + h * DKk, qs) * ATT_SCALE;
    float l1 = dot32_hq(kh + (rowb + o1) * DDd + h * DKk, qs) * ATT_SCALE;
    red[tid] = fmaxf(l0, l1); __syncthreads();
    for (int s = 128; s > 0; s >>= 1) { if (tid < s) red[tid] = fmaxf(red[tid], red[tid + s]); __syncthreads(); }
    const float mx = red[0]; __syncthreads();
    float a0 = l0 - mx, a1 = l1 - mx;
    if (msk[rowb + o0] == 0) a0 = -1e9f;
    if (msk[rowb + o1] == 0) a1 = -1e9f;
    red[tid] = fmaxf(a0, a1); __syncthreads();
    for (int s = 128; s > 0; s >>= 1) { if (tid < s) red[tid] = fmaxf(red[tid], red[tid + s]); __syncthreads(); }
    const float m2 = red[0]; __syncthreads();
    const float e0 = __expf(a0 - m2), e1 = __expf(a1 - m2);
    red[tid] = e0 + e1; __syncthreads();
    for (int s = 128; s > 0; s >>= 1) { if (tid < s) red[tid] += red[tid + s]; __syncthreads(); }
    const float isum = 1.0f / red[0]; __syncthreads();
    p[o0] = e0 * isum; p[o1] = e1 * isum;
    __syncthreads();
    const int dk = tid & 31, oc = tid >> 5;
    float acc = 0.f;
    for (int o = oc; o < OOo; o += 8)
      acc += p[o] * (float)vh[(rowb + o) * DDd + h * DKk + dk];
    part[oc][dk] = acc; __syncthreads();
    if (tid < 32) {
      float t = 0.f;
#pragma unroll
      for (int cc = 0; cc < 8; ++cc) t += part[cc][tid];
      res[(long)bm * DDd + h * DKk + tid] = t;
    }
    __syncthreads();
  }
}

// Operation-layer pass 1: per (b,h,m): logits[o], subtract max over o (dim=-1 quirk), mask.
__global__ __launch_bounds__(256) void hgan_op_logits(
    const float* __restrict__ q, const _Float16* __restrict__ kh,
    const int* __restrict__ msk, float* __restrict__ attb)
{
  __shared__ float red[256];
  const int idx = blockIdx.x;                // (b*H + h)*M + m
  const int b = idx >> 9;
  const int h = (idx >> 6) & 7;
  const int m = idx & 63;
  const int tid = threadIdx.x;
  const long arow = (long)(b * MMm + m) * OOo;
  const int o0 = tid, o1 = tid + 256;
  float q0v[32], q1v[32];
  {
    const v4f* q0p = (const v4f*)(q + (long)(b * OOo + o0) * DDd + h * DKk);
    const v4f* q1p = (const v4f*)(q + (long)(b * OOo + o1) * DDd + h * DKk);
#pragma unroll
    for (int i = 0; i < 8; ++i) {
      v4f u0 = q0p[i], u1 = q1p[i];
#pragma unroll
      for (int j = 0; j < 4; ++j) { q0v[i * 4 + j] = u0[j]; q1v[i * 4 + j] = u1[j]; }
    }
  }
  float l0 = dot32_hq(kh + (arow + o0) * DDd + h * DKk, q0v) * ATT_SCALE;
  float l1 = dot32_hq(kh + (arow + o1) * DDd + h * DKk, q1v) * ATT_SCALE;
  red[tid] = fmaxf(l0, l1); __syncthreads();
  for (int s = 128; s > 0; s >>= 1) { if (tid < s) red[tid] = fmaxf(red[tid], red[tid + s]); __syncthreads(); }
  const float mx = red[0];
  float a0 = l0 - mx, a1 = l1 - mx;
  if (msk[arow + o0] == 0) a0 = -1e9f;
  if (msk[arow + o1] == 0) a1 = -1e9f;
  attb[(long)idx * OOo + o0] = a0;
  attb[(long)idx * OOo + o1] = a1;
}

// Operation-layer pass 2: per (b,o): softmax over m (axis=2), then sum_m p*v.
__global__ __launch_bounds__(256) void hgan_op_attnout(
    const float* __restrict__ attb, const _Float16* __restrict__ vh,
    float* __restrict__ res)
{
  __shared__ float p[64];
  __shared__ float red[64];
  __shared__ float part[8][32];
  const int bo = blockIdx.x;                 // b*O + o
  const int b = bo >> 9, o = bo & 511;
  const int tid = threadIdx.x;
  for (int h = 0; h < HHh; ++h) {
    float a = -3.4e38f;
    if (tid < 64) { a = attb[(long)((b * HHh + h) * MMm + tid) * OOo + o]; red[tid] = a; }
    __syncthreads();
    for (int s = 32; s > 0; s >>= 1) { if (tid < s) red[tid] = fmaxf(red[tid], red[tid + s]); __syncthreads(); }
    const float mx = red[0]; __syncthreads();
    float e = 0.f;
    if (tid < 64) { e = __expf(a - mx); red[tid] = e; }
    __syncthreads();
    for (int s = 32; s > 0; s >>= 1) { if (tid < s) red[tid] += red[tid + s]; __syncthreads(); }
    const float isum = 1.0f / red[0]; __syncthreads();
    if (tid < 64) p[tid] = e * isum;
    __syncthreads();
    const int dk = tid & 31, mc = tid >> 5;
    float acc = 0.f;
#pragma unroll
    for (int mi = mc; mi < MMm; mi += 8)
      acc += p[mi] * (float)vh[((long)(b * MMm + mi) * OOo + o) * DDd + h * DKk + dk];
    part[mc][dk] = acc; __syncthreads();
    if (tid < 32) {
      float t = 0.f;
#pragma unroll
      for (int cc = 0; cc < 8; ++cc) t += part[cc][tid];
      res[(long)bo * DDd + h * DKk + tid] = t;
    }
    __syncthreads();
  }
}

static inline void run_gemm(hipStream_t st, _Float16* wt,
    const float* A, const float* invn, const float* cs, const float* W,
    const float* bias, float* outf, _Float16* outh,
    const float* r1, const float* r2,
    int R, int K, int N, int gelu,
    int batch = 1, long sA = 0, long sW = 0, long sOut = 0)
{
  const long kn = (long)K * N;
  dim3 pg((unsigned)((kn + 255) / 256), batch);
  hgan_prep_w<<<pg, 256, 0, st>>>(W, wt, K, N, sW);
  dim3 g(N / 64, R / 64, batch), blk(128, 1, 1);
  const long sWt = (batch > 1) ? kn : 0;
  const int cfg = (cs ? CFG_CS : 0) | (outh ? CFG_OUTH : 0) | (gelu ? CFG_GELU : 0)
                | (r1 ? CFG_R1 : 0) | (r2 ? CFG_R2 : 0) | (bias ? CFG_BIAS : 0);
#define HGAN_LG(C) hgan_wmma_gemm<C><<<g, blk, 0, st>>>(A, invn, cs, wt, bias, outf, outh, r1, r2, K, N, sA, sWt, sOut)
  switch (cfg) {
    case (CFG_CS | CFG_BIAS):                 HGAN_LG(CFG_CS | CFG_BIAS); break;
    case (CFG_CS | CFG_OUTH | CFG_BIAS):      HGAN_LG(CFG_CS | CFG_OUTH | CFG_BIAS); break;
    case (CFG_CS | CFG_GELU | CFG_BIAS):      HGAN_LG(CFG_CS | CFG_GELU | CFG_BIAS); break;
    case (CFG_R1 | CFG_BIAS):                 HGAN_LG(CFG_R1 | CFG_BIAS); break;
    case (CFG_R1 | CFG_R2 | CFG_BIAS):        HGAN_LG(CFG_R1 | CFG_R2 | CFG_BIAS); break;
    case (CFG_BIAS):                          HGAN_LG(CFG_BIAS); break;
    default:                                  HGAN_LG(0); break;   // adj gemm (no bias)
  }
#undef HGAN_LG
}

extern "C" void kernel_launch(void* const* d_in, const int* in_sizes, int n_in,
                              void* d_out, int out_size, void* d_ws, size_t ws_size,
                              hipStream_t stream) {
  (void)in_sizes; (void)n_in; (void)out_size; (void)ws_size;
  const float* h_m0  = (const float*)d_in[0];
  const float* h_o0  = (const float*)d_in[1];
  const float* h_arc = (const float*)d_in[2];
  const int*   mask  = (const int*)d_in[3];
  const int*   adj   = (const int*)d_in[4];

  // params flattened depth-first in dict insertion order
  int cur = 5;
  auto F = [&]() { return (const float*)d_in[cur++]; };
  struct AttnP { const float *wq, *bq, *wk, *bk, *wv, *bv, *wo, *bo; };
  struct FfnP  { const float *w1, *b1, *w2, *b2; };
  struct MachP { AttnP a; const float *n1, *n2, *n3; FfnP f; } mp[3];
  struct OperP { AttnP a; const float *n1, *n2, *n3, *n4, *wnext, *bnext; FfnP f; } op[3];
  for (int l = 0; l < 3; ++l) {
    mp[l].a = {F(), F(), F(), F(), F(), F(), F(), F()};
    mp[l].n1 = F(); mp[l].n2 = F(); mp[l].n3 = F();
    mp[l].f = {F(), F(), F(), F()};
    op[l].a = {F(), F(), F(), F(), F(), F(), F(), F()};
    op[l].n1 = F(); op[l].n2 = F(); op[l].n3 = F(); op[l].n4 = F();
    op[l].wnext = F(); op[l].bnext = F();
    op[l].f = {F(), F(), F(), F()};
  }
  const float* p_wm = F(); const float* p_bm = F();
  const float* p_wo = F(); const float* p_bo = F();

  // workspace carve-out
  char* base = (char*)d_ws; size_t off = 0;
  auto alloc = [&](size_t bytes) -> void* {
    void* pp = base + off; off += (bytes + 255) & ~(size_t)255; return pp;
  };
  float* hm      = (float*)alloc((size_t)BB * MMm * DDd * 4);   // evolving h_m
  float* ho      = (float*)alloc((size_t)BB * OOo * DDd * 4);   // evolving h_o
  float* inv_m   = (float*)alloc((size_t)BB * MMm * 4);
  float* inv_o   = (float*)alloc((size_t)BB * OOo * 4);
  float* inv_arc = (float*)alloc((size_t)RARC * 4);
  float* qb      = (float*)alloc((size_t)BB * OOo * DDd * 4);
  float* resb    = (float*)alloc((size_t)BB * OOo * DDd * 4);
  float* hidb    = (float*)alloc((size_t)BB * OOo * DDd * 4);   // covers 512x512 machine hid
  float* ub      = (float*)alloc((size_t)BB * OOo * DDd * 4);
  float* hnx     = (float*)alloc((size_t)BB * OOo * DDd * 4);
  float* attb    = (float*)alloc((size_t)BB * HHh * MMm * OOo * 4);
  float* adjf    = (float*)alloc((size_t)BB * OOo * OOo * 4);
  _Float16* wt   = (_Float16*)alloc((size_t)BB * OOo * DDd * 2);  // transposed-weight scratch
  _Float16* kh   = (_Float16*)alloc((size_t)RARC * DDd * 2);
  _Float16* vh   = (_Float16*)alloc((size_t)RARC * DDd * 2);

  const int RM = BB * MMm;   // 512
  const int RO = BB * OOo;   // 4096

  hipMemcpyAsync(hm, h_m0, (size_t)RM * DDd * 4, hipMemcpyDeviceToDevice, stream);
  hipMemcpyAsync(ho, h_o0, (size_t)RO * DDd * 4, hipMemcpyDeviceToDevice, stream);
  hgan_i2f<<<(BB * OOo * OOo + 255) / 256, 256, 0, stream>>>(adj, adjf, BB * OOo * OOo);
  // h_arc is constant across layers: its row norms are computed exactly once.
  hgan_rowinv<<<RARC / 8, 256, 0, stream>>>(h_arc, inv_arc, RARC, DDd);

  for (int l = 0; l < 3; ++l) {
    const MachP& P = mp[l];
    // ---- machine layer ----
    hgan_rowinv<<<RM / 8, 256, 0, stream>>>(hm, inv_m, RM, DDd);
    run_gemm(stream, wt, hm, inv_m, P.n1, P.a.wq, P.a.bq, qb, nullptr, nullptr, nullptr, RM, DDd, DDd, 0);
    run_gemm(stream, wt, h_arc, inv_arc, P.n2, P.a.wk, P.a.bk, nullptr, kh, nullptr, nullptr, RARC, DDd, DDd, 0);
    run_gemm(stream, wt, h_arc, inv_arc, P.n2, P.a.wv, P.a.bv, nullptr, vh, nullptr, nullptr, RARC, DDd, DDd, 0);
    hgan_attn_machine<<<RM, 256, 0, stream>>>(qb, kh, vh, mask, resb);
    run_gemm(stream, wt, resb, nullptr, nullptr, P.a.wo, P.a.bo, hm, nullptr, hm, nullptr, RM, DDd, DDd, 0);
    hgan_rowinv<<<RM / 8, 256, 0, stream>>>(hm, inv_m, RM, DDd);
    run_gemm(stream, wt, hm, inv_m, P.n3, P.f.w1, P.f.b1, hidb, nullptr, nullptr, nullptr, RM, DDd, 512, 1);
    run_gemm(stream, wt, hidb, nullptr, nullptr, P.f.w2, P.f.b2, hm, nullptr, hm, nullptr, RM, 512, DDd, 0);

    const OperP& Q = op[l];
    // ---- operation layer ----
    hgan_rowinv<<<RO / 8, 256, 0, stream>>>(ho, inv_o, RO, DDd);   // shared by n4 and n1 paths
    run_gemm(stream, wt, ho, inv_o, Q.n4, Q.wnext, Q.bnext, ub, nullptr, nullptr, nullptr, RO, DDd, DDd, 0);
    // h_next[b] = adj[b] (512x512) @ u[b] (512x256), batched over b
    run_gemm(stream, wt, adjf, nullptr, nullptr, ub, nullptr, hnx, nullptr, nullptr, nullptr,
             OOo, OOo, DDd, 0, BB, (long)OOo * OOo, (long)OOo * DDd, (long)OOo * DDd);
    run_gemm(stream, wt, ho, inv_o, Q.n1, Q.a.wq, Q.a.bq, qb, nullptr, nullptr, nullptr, RO, DDd, DDd, 0);
    run_gemm(stream, wt, h_arc, inv_arc, Q.n2, Q.a.wk, Q.a.bk, nullptr, kh, nullptr, nullptr, RARC, DDd, DDd, 0);
    run_gemm(stream, wt, h_arc, inv_arc, Q.n2, Q.a.wv, Q.a.bv, nullptr, vh, nullptr, nullptr, RARC, DDd, DDd, 0);
    hgan_op_logits<<<BB * HHh * MMm, 256, 0, stream>>>(qb, kh, mask, attb);
    hgan_op_attnout<<<RO, 256, 0, stream>>>(attb, vh, resb);
    run_gemm(stream, wt, resb, nullptr, nullptr, Q.a.wo, Q.a.bo, ho, nullptr, ho, hnx, RO, DDd, DDd, 0);
    hgan_rowinv<<<RO / 8, 256, 0, stream>>>(ho, inv_o, RO, DDd);
    run_gemm(stream, wt, ho, inv_o, Q.n3, Q.f.w1, Q.f.b1, hidb, nullptr, nullptr, nullptr, RO, DDd, DDd, 1);
    run_gemm(stream, wt, hidb, nullptr, nullptr, Q.f.w2, Q.f.b2, ho, nullptr, ho, nullptr, RO, DDd, DDd, 0);
  }

  // outputs: (out_o, out_m) concatenated
  float* out_o = (float*)d_out;
  float* out_m = out_o + (size_t)RO * DDd;
  run_gemm(stream, wt, ho, nullptr, nullptr, p_wo, p_bo, out_o, nullptr, nullptr, nullptr, RO, DDd, DDd, 0);
  run_gemm(stream, wt, hm, nullptr, nullptr, p_wm, p_bm, out_m, nullptr, nullptr, nullptr, RM, DDd, DDd, 0);
}
